// STMambaBlock_2731599200646
// MI455X (gfx1250) — compile-verified, matrix operand order checked by reading, not run on hardware
//
#include <hip/hip_runtime.h>

// ---------------- constants ----------------
#define Bc 2
#define Tc 32
#define NODEc 307
#define Dc 128
#define DINNER 256
#define DSTATE 16
#define DTRANK 8
#define DFF 512
#define NEXP 4
#define TOKc (Bc*Tc*NODEc)   // 19648
#define MSEQc (Bc*NODEc)     // 614
#define XDW 48               // padded 40 -> 48
#define NPAD 320             // padded 307 -> 320

typedef __attribute__((ext_vector_type(16))) __bf16 bfx16;
typedef __attribute__((ext_vector_type(8)))  float  fx8;

union BFrag { bfx16 v; uint4 q[2]; };

__device__ __forceinline__ __bf16 f2bf(float x) {
  unsigned u = __builtin_bit_cast(unsigned, x);
  unsigned r = u + 0x7FFFu + ((u >> 16) & 1u);
  unsigned short h = (unsigned short)(r >> 16);
  return __builtin_bit_cast(__bf16, h);
}
__device__ __forceinline__ float gelu_f(float x) {
  return 0.5f * x * (1.0f + erff(x * 0.70710678118654752f));
}
__device__ __forceinline__ float silu_f(float x) {
  return x / (1.0f + __expf(-x));
}

// Fragment loader: two contiguous 16B chunks per lane (works for A rows and W rows).
__device__ __forceinline__ bfx16 load_frag(const __bf16* __restrict__ P, long row, int ld, int kb) {
  const __bf16* p = P + row * (long)ld + kb;
  BFrag f;
  f.q[0] = *reinterpret_cast<const uint4*>(p);
  f.q[1] = *reinterpret_cast<const uint4*>(p + 16);
  return f.v;
}

// ---------------- generic WMMA GEMM: out = A(bf16, MxK) * W(bf16, NxK)^T ----------------
// Wave tile 32x64 (2 M-tiles x 4 N-tiles, 8 WMMA per k-step); block = 4 waves = 64x128.
// flags bit0: accumulate into C (fp32); bit1: exact GELU before store.
// C (fp32) and CB (bf16) are each optional outputs.
__global__ __launch_bounds__(128) void gemm_bf16_kernel(
    const __bf16* __restrict__ A, const __bf16* __restrict__ W,
    float* __restrict__ C, __bf16* __restrict__ CB,
    int M, int N, int K, int lda, int ldb, int ldc,
    long long sA, long long sW, long long sC, int flags)
{
  A += (long long)blockIdx.z * sA;
  W += (long long)blockIdx.z * sW;
  if (C)  C  += (long long)blockIdx.z * sC;
  if (CB) CB += (long long)blockIdx.z * sC;

  const int lane = threadIdx.x & 31;
  const int wave = threadIdx.x >> 5;
  const int l16 = lane & 15, hi = lane >> 4;
  const int mbase = blockIdx.y * 64 + (wave >> 1) * 32;
  const int nbase = blockIdx.x * 128 + (wave & 1) * 64;

  fx8 acc[2][4];
  #pragma unroll
  for (int i = 0; i < 2; i++)
    #pragma unroll
    for (int j = 0; j < 4; j++)
      #pragma unroll
      for (int q = 0; q < 8; q++) acc[i][j][q] = 0.0f;

  long arow[2];
  #pragma unroll
  for (int tm = 0; tm < 2; tm++) {
    long r = mbase + tm * 16 + l16;
    if (r > (long)M - 1) r = (long)M - 1;
    arow[tm] = r;
  }
  int nrow[4]; bool nok[4];
  #pragma unroll
  for (int tn = 0; tn < 4; tn++) {
    int nb = nbase + tn * 16;
    nok[tn] = nb < N;
    nrow[tn] = nok[tn] ? (nb + l16) : 0;
  }

  for (int k0 = 0; k0 < K; k0 += 32) {
    int kb = k0 + hi * 8;
    bfx16 af[2];
    #pragma unroll
    for (int tm = 0; tm < 2; tm++) af[tm] = load_frag(A, arow[tm], lda, kb);
    bfx16 bfv[4];
    #pragma unroll
    for (int tn = 0; tn < 4; tn++) bfv[tn] = load_frag(W, nrow[tn], ldb, kb);
    #pragma unroll
    for (int tm = 0; tm < 2; tm++)
      #pragma unroll
      for (int tn = 0; tn < 4; tn++)
        acc[tm][tn] = __builtin_amdgcn_wmma_f32_16x16x32_bf16(
            false, af[tm], false, bfv[tn], (short)0, acc[tm][tn], false, false);
  }

  #pragma unroll
  for (int tm = 0; tm < 2; tm++)
    #pragma unroll
    for (int tn = 0; tn < 4; tn++) {
      if (!nok[tn]) continue;
      int col = nbase + tn * 16 + l16;
      #pragma unroll
      for (int i = 0; i < 8; i++) {
        int row = mbase + tm * 16 + hi * 8 + i;
        if (row < M) {
          long off = (long)row * ldc + col;
          float v = acc[tm][tn][i];
          if (flags & 2) v = gelu_f(v);
          if (flags & 1) v += C[off];
          if (C)  C[off] = v;
          if (CB) CB[off] = f2bf(v);
        }
      }
    }
}

// ---------------- weight f32 -> bf16 (with padding / column slice) ----------------
__global__ void convert_bf_kernel(const float* __restrict__ src, __bf16* __restrict__ dst,
                                  int rows_out, int cols_out, int rows_src, int cols_src,
                                  int src_ld, int col_off)
{
  long idx = (long)blockIdx.x * 256 + threadIdx.x;
  long total = (long)rows_out * cols_out;
  if (idx >= total) return;
  int c = (int)(idx % cols_out);
  long r = idx / cols_out;
  float v = 0.0f;
  if (r < rows_src && c < cols_src) v = src[r * (long)src_ld + col_off + c];
  dst[idx] = f2bf(v);
}

// ---------------- layout shuffles ----------------
__global__ void gather_xr_kernel(const float* __restrict__ x, float* __restrict__ XR)
{
  long row = blockIdx.x; int d = threadIdx.x;          // row = (b*NODE+n)*T + t
  int t = (int)(row & 31); long m = row >> 5;
  int b = (int)(m / NODEc); int n = (int)(m % NODEc);
  long src = ((long)(b * Tc + t) * NODEc + n);
  XR[row * Dc + d] = x[src * Dc + d];
}

__global__ void scatter_out_kernel(const float* __restrict__ XR, float* __restrict__ out)
{
  long row = blockIdx.x; int d = threadIdx.x;          // row = (b*T+t)*NODE + n
  int n = (int)(row % NODEc); long g = row / NODEc;
  int t = (int)(g & 31); int b = (int)(g >> 5);
  long src = ((long)(b * NODEc + n) * Tc + t);
  out[row * Dc + d] = XR[src * Dc + d];
}

__global__ void flip_bf16_kernel(const __bf16* __restrict__ in, __bf16* __restrict__ out)
{
  long row = blockIdx.x; int d = threadIdx.x;
  long m = row >> 5; int t = (int)(row & 31);
  out[row * Dc + d] = in[(m * Tc + (Tc - 1 - t)) * Dc + d];
}

__global__ void add_inplace_kernel(float* __restrict__ dst, const float* __restrict__ src, long total)
{
  long i = (long)blockIdx.x * 256 + threadIdx.x;
  if (i < total) dst[i] += src[i];
}

// ---------------- RMSNorm (one wave32 per 128-wide row; fp32 + bf16 outputs) ----------------
__global__ __launch_bounds__(256) void rmsnorm_kernel(const float* __restrict__ X,
                                                      const float* __restrict__ w,
                                                      float* __restrict__ O,
                                                      __bf16* __restrict__ OB,
                                                      int rows)
{
  int wave = threadIdx.x >> 5, lane = threadIdx.x & 31;
  long row = (long)blockIdx.x * 8 + wave;
  if (row >= rows) return;
  float4 v = reinterpret_cast<const float4*>(X + row * Dc)[lane];
  float s = v.x * v.x + v.y * v.y + v.z * v.z + v.w * v.w;
  #pragma unroll
  for (int off = 16; off > 0; off >>= 1) s += __shfl_xor(s, off, 32);
  float sc = rsqrtf(s * (1.0f / Dc) + 1e-6f);
  float4 g = reinterpret_cast<const float4*>(w)[lane];
  float4 o; o.x = v.x * sc * g.x; o.y = v.y * sc * g.y; o.z = v.z * sc * g.z; o.w = v.w * sc * g.w;
  if (O) reinterpret_cast<float4*>(O + row * Dc)[lane] = o;
  __bf16* ob = OB + row * Dc + lane * 4;
  ob[0] = f2bf(o.x); ob[1] = f2bf(o.y); ob[2] = f2bf(o.z); ob[3] = f2bf(o.w);
}

// ---------------- causal depthwise conv1d + SiLU (fp32 for scan, bf16 for GEMM) ----------------
__global__ __launch_bounds__(256) void conv_silu_kernel(const float* __restrict__ XZ,
                                                        const float* __restrict__ cw,
                                                        const float* __restrict__ cb,
                                                        float* __restrict__ XC,
                                                        __bf16* __restrict__ XCB, int rev)
{
  int m = blockIdx.x, c = threadIdx.x;
  float w0 = cw[c * 4 + 0], w1 = cw[c * 4 + 1], w2 = cw[c * 4 + 2], w3 = cw[c * 4 + 3];
  float b = cb[c];
  float x0 = 0.f, x1 = 0.f, x2 = 0.f;
  for (int t = 0; t < Tc; ++t) {
    long srow = (long)m * Tc + (rev ? (Tc - 1 - t) : t);
    float xt = XZ[srow * (2 * DINNER) + c];
    float a = b + w0 * x0 + w1 * x1 + w2 * x2 + w3 * xt;
    float s = silu_f(a);
    long o = ((long)m * Tc + t) * DINNER + c;
    XC[o] = s;
    XCB[o] = f2bf(s);
    x0 = x1; x1 = x2; x2 = xt;
  }
}

// ---------------- dt projection + softplus ----------------
__global__ __launch_bounds__(256) void dtproj_kernel(const float* __restrict__ XD,
                                                     const float* __restrict__ w_dt,
                                                     const float* __restrict__ b_dt,
                                                     float* __restrict__ DT)
{
  long row = blockIdx.x; int c = threadIdx.x;
  const float* xd = XD + row * XDW;
  float v = b_dt[c];
  #pragma unroll
  for (int r = 0; r < DTRANK; ++r) v += xd[r] * w_dt[c * DTRANK + r];
  DT[row * DINNER + c] = (v > 20.0f) ? v : log1pf(__expf(v));
}

// ---------------- selective scan + SiLU gate -> bf16 (feeds out-proj GEMM) ----------------
__global__ __launch_bounds__(256) void scan_kernel(const float* __restrict__ XZ,
                                                   const float* __restrict__ XD,
                                                   const float* __restrict__ DT,
                                                   const float* __restrict__ XC,
                                                   __bf16* __restrict__ Y,
                                                   const float* __restrict__ a_log,
                                                   const float* __restrict__ Dp, int rev)
{
  int m = blockIdx.x, c = threadIdx.x;
  float Ac[DSTATE], h[DSTATE];
  #pragma unroll
  for (int s = 0; s < DSTATE; ++s) { Ac[s] = -__expf(a_log[c * DSTATE + s]); h[s] = 0.f; }
  float Dv = Dp[c];
  for (int t = 0; t < Tc; ++t) {
    long row = (long)m * Tc + t;
    float u  = XC[row * DINNER + c];
    float dt = DT[row * DINNER + c];
    const float* xd = XD + row * XDW;
    float y = 0.f;
    #pragma unroll
    for (int s = 0; s < DSTATE; ++s) {
      float dA = __expf(dt * Ac[s]);
      h[s] = dA * h[s] + dt * xd[DTRANK + s] * u;
      y += h[s] * xd[DTRANK + DSTATE + s];
    }
    y += u * Dv;
    long zrow = (long)m * Tc + (rev ? (Tc - 1 - t) : t);
    float z = XZ[zrow * (2 * DINNER) + DINNER + c];
    Y[row * DINNER + c] = f2bf(y * silu_f(z));
  }
}

// ---------------- adjacency: softmax(E E^T / 4) -> bf16, K-padded to 320 ----------------
__global__ __launch_bounds__(128) void adj_kernel(const float* __restrict__ E, __bf16* __restrict__ adj)
{
  __shared__ float l[NPAD];
  __shared__ float red[128];
  int r = blockIdx.x, tid = threadIdx.x;
  float er[16];
  #pragma unroll
  for (int j = 0; j < 16; ++j) er[j] = E[r * 16 + j];
  for (int n = tid; n < NPAD; n += 128) {
    float d = 0.f;
    if (n < NODEc) {
      #pragma unroll
      for (int j = 0; j < 16; ++j) d += er[j] * E[n * 16 + j];
      l[n] = d * 0.25f;
    } else l[n] = -1e30f;
  }
  __syncthreads();
  float mx = -1e30f;
  for (int n = tid; n < NODEc; n += 128) mx = fmaxf(mx, l[n]);
  red[tid] = mx; __syncthreads();
  for (int s = 64; s > 0; s >>= 1) { if (tid < s) red[tid] = fmaxf(red[tid], red[tid + s]); __syncthreads(); }
  float rmax = red[0]; __syncthreads();
  float sm = 0.f;
  for (int n = tid; n < NODEc; n += 128) sm += __expf(l[n] - rmax);
  red[tid] = sm; __syncthreads();
  for (int s = 64; s > 0; s >>= 1) { if (tid < s) red[tid] += red[tid + s]; __syncthreads(); }
  float inv = 1.0f / red[0];
  for (int n = tid; n < NPAD; n += 128)
    adj[(long)r * NPAD + n] = f2bf((n < NODEc) ? __expf(l[n] - rmax) * inv : 0.0f);
}

// ---------------- msg^T to bf16, per group, K-padded (n: 307 -> 320) ----------------
__global__ void msgt_kernel(const float* __restrict__ MSG, __bf16* __restrict__ MSGT)
{
  long idx = (long)blockIdx.x * 256 + threadIdx.x;
  long total = (long)(Bc * Tc) * Dc * NPAD;
  if (idx >= total) return;
  int n = (int)(idx % NPAD);
  long r = idx / NPAD;
  int d = (int)(r % Dc);
  int g = (int)(r / Dc);
  float v = 0.f;
  if (n < NODEc) v = MSG[((long)g * NODEc + n) * Dc + d];
  MSGT[idx] = f2bf(v);
}

// ---------------- router: softmax over 4 experts, top-2 renormalized gates ----------------
__global__ void router_kernel(const float* __restrict__ XN, const float* __restrict__ wr,
                              float* __restrict__ G, int tok)
{
  int t = blockIdx.x * 256 + threadIdx.x;
  if (t >= tok) return;
  const float* x = XN + (long)t * Dc;
  float lg[NEXP];
  #pragma unroll
  for (int e = 0; e < NEXP; ++e) {
    float s = 0.f;
    for (int d = 0; d < Dc; ++d) s += x[d] * wr[e * Dc + d];
    lg[e] = s;
  }
  float mx = lg[0];
  #pragma unroll
  for (int e = 1; e < NEXP; ++e) mx = fmaxf(mx, lg[e]);
  float p[NEXP], sum = 0.f;
  #pragma unroll
  for (int e = 0; e < NEXP; ++e) { p[e] = __expf(lg[e] - mx); sum += p[e]; }
  #pragma unroll
  for (int e = 0; e < NEXP; ++e) p[e] /= sum;
  int i1 = 0;
  #pragma unroll
  for (int e = 1; e < NEXP; ++e) if (p[e] > p[i1]) i1 = e;
  int i2 = -1;
  #pragma unroll
  for (int e = 0; e < NEXP; ++e) { if (e == i1) continue; if (i2 < 0 || p[e] > p[i2]) i2 = e; }
  float s2 = p[i1] + p[i2];
  #pragma unroll
  for (int e = 0; e < NEXP; ++e)
    G[(long)t * NEXP + e] = (e == i1) ? p[i1] / s2 : ((e == i2) ? p[i2] / s2 : 0.0f);
}

__global__ void combine_kernel(const float* __restrict__ YE, const float* __restrict__ G,
                               float* __restrict__ out, int e, long total)
{
  long i = (long)blockIdx.x * 256 + threadIdx.x;
  if (i >= total) return;
  long t = i >> 7;
  out[i] += G[t * NEXP + e] * YE[i];
}

// ---------------- host orchestration ----------------
static inline void launch_gemm(hipStream_t s, const __bf16* A, const __bf16* W,
                               float* C, __bf16* CB,
                               int M, int N, int K, int lda, int ldb, int ldc,
                               int batches, long long sA, long long sW, long long sC, int flags)
{
  dim3 g((unsigned)((N + 127) / 128), (unsigned)((M + 63) / 64), (unsigned)batches);
  gemm_bf16_kernel<<<g, 128, 0, s>>>(A, W, C, CB, M, N, K, lda, ldb, ldc, sA, sW, sC, flags);
}

extern "C" void kernel_launch(void* const* d_in, const int* in_sizes, int n_in,
                              void* d_out, int out_size, void* d_ws, size_t ws_size,
                              hipStream_t stream)
{
  (void)in_sizes; (void)n_in; (void)out_size; (void)ws_size;
  const int TOK = TOKc;

  const float* x        = (const float*)d_in[0];
  const float* w_tnorm  = (const float*)d_in[1];
  const float* f_w_in   = (const float*)d_in[2];
  const float* f_convw  = (const float*)d_in[3];
  const float* f_convb  = (const float*)d_in[4];
  const float* f_w_x    = (const float*)d_in[5];
  const float* f_w_dt   = (const float*)d_in[6];
  const float* f_b_dt   = (const float*)d_in[7];
  const float* f_alog   = (const float*)d_in[8];
  const float* f_d      = (const float*)d_in[9];
  const float* f_w_out  = (const float*)d_in[10];
  const float* b_w_in   = (const float*)d_in[11];
  const float* b_convw  = (const float*)d_in[12];
  const float* b_convb  = (const float*)d_in[13];
  const float* b_w_x    = (const float*)d_in[14];
  const float* b_w_dt   = (const float*)d_in[15];
  const float* b_b_dt   = (const float*)d_in[16];
  const float* b_alog   = (const float*)d_in[17];
  const float* b_d      = (const float*)d_in[18];
  const float* b_w_out  = (const float*)d_in[19];
  const float* w_merge  = (const float*)d_in[20];
  const float* w_snorm  = (const float*)d_in[21];
  const float* node_e   = (const float*)d_in[22];
  const float* w_msg    = (const float*)d_in[23];
  const float* w_upd    = (const float*)d_in[24];
  const float* w_fnorm  = (const float*)d_in[25];
  const float* w_router = (const float*)d_in[26];
  const float* w1       = (const float*)d_in[27];
  const float* w2       = (const float*)d_in[28];

  char* wsp = (char*)d_ws;
  auto alloc = [&](size_t elems, size_t esz) -> void* {
    void* p = (void*)wsp;
    size_t bytes = (elems * esz + 255) & ~(size_t)255;
    wsp += bytes;
    return p;
  };
  // fp32 activations
  float* XR   = (float*)alloc((size_t)TOK * Dc, 4);
  float* XN   = (float*)alloc((size_t)TOK * Dc, 4);
  float* XZf  = (float*)alloc((size_t)TOK * 2 * DINNER, 4);
  float* XZb  = (float*)alloc((size_t)TOK * 2 * DINNER, 4);
  float* XCf  = (float*)alloc((size_t)TOK * DINNER, 4);
  float* XCb  = (float*)alloc((size_t)TOK * DINNER, 4);
  float* XDf  = (float*)alloc((size_t)TOK * XDW, 4);
  float* XDb  = (float*)alloc((size_t)TOK * XDW, 4);
  float* DTf  = (float*)alloc((size_t)TOK * DINNER, 4);
  float* DTb  = (float*)alloc((size_t)TOK * DINNER, 4);
  float* MRG  = (float*)alloc((size_t)TOK * Dc, 4);
  float* MSG  = (float*)alloc((size_t)TOK * Dc, 4);
  float* YE   = (float*)alloc((size_t)TOK * Dc, 4);
  float* GT   = (float*)alloc((size_t)TOK * NEXP, 4);
  // bf16 activations (GEMM operands)
  __bf16* XNb   = (__bf16*)alloc((size_t)TOK * Dc, 2);
  __bf16* XCfb  = (__bf16*)alloc((size_t)TOK * DINNER, 2);
  __bf16* XCbb  = (__bf16*)alloc((size_t)TOK * DINNER, 2);
  __bf16* Yfb   = (__bf16*)alloc((size_t)TOK * DINNER, 2);
  __bf16* Ybb   = (__bf16*)alloc((size_t)TOK * DINNER, 2);
  __bf16* OFb   = (__bf16*)alloc((size_t)TOK * Dc, 2);
  __bf16* OBb   = (__bf16*)alloc((size_t)TOK * Dc, 2);
  __bf16* OBrb  = (__bf16*)alloc((size_t)TOK * Dc, 2);
  __bf16* ADJb  = (__bf16*)alloc((size_t)NODEc * NPAD, 2);
  __bf16* AGGb  = (__bf16*)alloc((size_t)TOK * Dc, 2);
  __bf16* Hb    = (__bf16*)alloc((size_t)TOK * DFF, 2);
  __bf16* MSGT  = (__bf16*)alloc((size_t)(Bc * Tc) * Dc * NPAD, 2);
  // bf16 weights
  __bf16* bw_in_f  = (__bf16*)alloc((size_t)2 * DINNER * Dc, 2);
  __bf16* bw_in_b  = (__bf16*)alloc((size_t)2 * DINNER * Dc, 2);
  __bf16* bw_x_f   = (__bf16*)alloc((size_t)XDW * DINNER, 2);
  __bf16* bw_x_b   = (__bf16*)alloc((size_t)XDW * DINNER, 2);
  __bf16* bw_out_f = (__bf16*)alloc((size_t)Dc * DINNER, 2);
  __bf16* bw_out_b = (__bf16*)alloc((size_t)Dc * DINNER, 2);
  __bf16* bWm_a    = (__bf16*)alloc((size_t)Dc * Dc, 2);
  __bf16* bWm_b    = (__bf16*)alloc((size_t)Dc * Dc, 2);
  __bf16* bw_msg   = (__bf16*)alloc((size_t)Dc * Dc, 2);
  __bf16* bw_upd   = (__bf16*)alloc((size_t)Dc * Dc, 2);
  __bf16* bw1      = (__bf16*)alloc((size_t)NEXP * DFF * Dc, 2);
  __bf16* bw2      = (__bf16*)alloc((size_t)NEXP * Dc * DFF, 2);

  float* OUT = (float*)d_out;

  auto cvt = [&](const float* src, __bf16* dst, int ro, int co, int rs, int cs, int ld, int off) {
    long total = (long)ro * co;
    convert_bf_kernel<<<dim3((unsigned)((total + 255) / 256)), 256, 0, stream>>>(src, dst, ro, co, rs, cs, ld, off);
  };

  // ---- weight conversions (fp32 -> bf16, padded/sliced) ----
  cvt(f_w_in, bw_in_f, 2 * DINNER, Dc, 2 * DINNER, Dc, Dc, 0);
  cvt(b_w_in, bw_in_b, 2 * DINNER, Dc, 2 * DINNER, Dc, Dc, 0);
  cvt(f_w_x, bw_x_f, XDW, DINNER, 40, DINNER, DINNER, 0);
  cvt(b_w_x, bw_x_b, XDW, DINNER, 40, DINNER, DINNER, 0);
  cvt(f_w_out, bw_out_f, Dc, DINNER, Dc, DINNER, DINNER, 0);
  cvt(b_w_out, bw_out_b, Dc, DINNER, Dc, DINNER, DINNER, 0);
  cvt(w_merge, bWm_a, Dc, Dc, Dc, Dc, 2 * Dc, 0);
  cvt(w_merge, bWm_b, Dc, Dc, Dc, Dc, 2 * Dc, Dc);
  cvt(w_msg, bw_msg, Dc, Dc, Dc, Dc, Dc, 0);
  cvt(w_upd, bw_upd, Dc, Dc, Dc, Dc, Dc, 0);
  cvt(w1, bw1, NEXP * DFF, Dc, NEXP * DFF, Dc, Dc, 0);
  cvt(w2, bw2, NEXP * Dc, DFF, NEXP * Dc, DFF, DFF, 0);

  // ---- temporal branch ----
  gather_xr_kernel<<<TOK, Dc, 0, stream>>>(x, XR);
  rmsnorm_kernel<<<TOK / 8, 256, 0, stream>>>(XR, w_tnorm, (float*)nullptr, XNb, TOK);

  launch_gemm(stream, XNb, bw_in_f, XZf, nullptr, TOK, 2 * DINNER, Dc, Dc, Dc, 2 * DINNER, 1, 0, 0, 0, 0);
  launch_gemm(stream, XNb, bw_in_b, XZb, nullptr, TOK, 2 * DINNER, Dc, Dc, Dc, 2 * DINNER, 1, 0, 0, 0, 0);

  conv_silu_kernel<<<MSEQc, DINNER, 0, stream>>>(XZf, f_convw, f_convb, XCf, XCfb, 0);
  conv_silu_kernel<<<MSEQc, DINNER, 0, stream>>>(XZb, b_convw, b_convb, XCb, XCbb, 1);

  launch_gemm(stream, XCfb, bw_x_f, XDf, nullptr, TOK, XDW, DINNER, DINNER, DINNER, XDW, 1, 0, 0, 0, 0);
  launch_gemm(stream, XCbb, bw_x_b, XDb, nullptr, TOK, XDW, DINNER, DINNER, DINNER, XDW, 1, 0, 0, 0, 0);

  dtproj_kernel<<<TOK, DINNER, 0, stream>>>(XDf, f_w_dt, f_b_dt, DTf);
  dtproj_kernel<<<TOK, DINNER, 0, stream>>>(XDb, b_w_dt, b_b_dt, DTb);

  scan_kernel<<<MSEQc, DINNER, 0, stream>>>(XZf, XDf, DTf, XCf, Yfb, f_alog, f_d, 0);
  scan_kernel<<<MSEQc, DINNER, 0, stream>>>(XZb, XDb, DTb, XCb, Ybb, b_alog, b_d, 1);

  launch_gemm(stream, Yfb, bw_out_f, nullptr, OFb, TOK, Dc, DINNER, DINNER, DINNER, Dc, 1, 0, 0, 0, 0);
  launch_gemm(stream, Ybb, bw_out_b, nullptr, OBb, TOK, Dc, DINNER, DINNER, DINNER, Dc, 1, 0, 0, 0, 0);

  flip_bf16_kernel<<<TOK, Dc, 0, stream>>>(OBb, OBrb);
  launch_gemm(stream, OFb, bWm_a, MRG, nullptr, TOK, Dc, Dc, Dc, Dc, Dc, 1, 0, 0, 0, 0);
  launch_gemm(stream, OBrb, bWm_b, MRG, nullptr, TOK, Dc, Dc, Dc, Dc, Dc, 1, 0, 0, 0, 1);
  add_inplace_kernel<<<(unsigned)(((long)TOK * Dc + 255) / 256), 256, 0, stream>>>(XR, MRG, (long)TOK * Dc);

  // ---- spatial branch (result accumulates directly into d_out) ----
  scatter_out_kernel<<<TOK, Dc, 0, stream>>>(XR, OUT);
  rmsnorm_kernel<<<TOK / 8, 256, 0, stream>>>(OUT, w_snorm, (float*)nullptr, XNb, TOK);
  adj_kernel<<<NODEc, 128, 0, stream>>>(node_e, ADJb);
  launch_gemm(stream, XNb, bw_msg, MSG, nullptr, TOK, Dc, Dc, Dc, Dc, Dc, 1, 0, 0, 0, 0);
  msgt_kernel<<<(unsigned)(((long)(Bc * Tc) * Dc * NPAD + 255) / 256), 256, 0, stream>>>(MSG, MSGT);
  launch_gemm(stream, ADJb, MSGT, nullptr, AGGb, NODEc, Dc, NPAD, NPAD, NPAD, Dc,
              Bc * Tc, 0, (long long)Dc * NPAD, (long long)NODEc * Dc, 0);
  launch_gemm(stream, AGGb, bw_upd, OUT, nullptr, TOK, Dc, Dc, Dc, Dc, Dc, 1, 0, 0, 0, 1);

  // ---- MoE FFN ----
  rmsnorm_kernel<<<TOK / 8, 256, 0, stream>>>(OUT, w_fnorm, XN, XNb, TOK);
  router_kernel<<<(TOK + 255) / 256, 256, 0, stream>>>(XN, w_router, GT, TOK);
  for (int e = 0; e < NEXP; ++e) {
    launch_gemm(stream, XNb, bw1 + (long)e * DFF * Dc, nullptr, Hb, TOK, DFF, Dc, Dc, Dc, DFF, 1, 0, 0, 0, 2);
    launch_gemm(stream, Hb, bw2 + (long)e * Dc * DFF, YE, nullptr, TOK, Dc, DFF, DFF, DFF, Dc, 1, 0, 0, 0, 0);
    combine_kernel<<<(unsigned)(((long)TOK * Dc + 255) / 256), 256, 0, stream>>>(YE, GT, OUT, e, (long)TOK * Dc);
  }
}
